// RoPEMultiHeadAttention_49632642072687
// MI455X (gfx1250) — matrix operand order repaired and checked by reading, not compiled, not run on hardware
//
#include <hip/hip_runtime.h>
#include <hip/hip_bf16.h>
#include <math.h>
#include <stdint.h>

#define DMODEL 1024
#define NHEAD  16
#define HDIM   64
#define BB     2
#define SS     2048

typedef __attribute__((ext_vector_type(16))) __bf16 v16bf;
typedef __attribute__((ext_vector_type(8)))  __bf16 v8bf;
typedef __attribute__((ext_vector_type(8)))  float  v8f;

// ---------------------------------------------------------------------------
// WMMA fragment loaders (gfx1250 wave32 layouts, 16x16x32 bf16)
// A (16x32): lane m = L&15; k chunks {kb..kb+7, kb+16..kb+23}, kb = (L<16?0:8)
// B (32x16): lane n = L&15; k chunks {kb..kb+7, kb+16..kb+23}, kb = (L<16?0:8)
// ---------------------------------------------------------------------------
__device__ __forceinline__ v16bf ld_a_frag(const __bf16* base, int ld, int m0, int k0) {
  const int lane = threadIdx.x & 31;
  const int m  = m0 + (lane & 15);
  const int kb = k0 + ((lane & 16) ? 8 : 0);
  const __bf16* p = base + (size_t)m * ld + kb;
  v8bf lo = *(const v8bf*)(p);
  v8bf hi = *(const v8bf*)(p + 16);
  v16bf r;
#pragma unroll
  for (int i = 0; i < 8; ++i) { r[i] = lo[i]; r[i + 8] = hi[i]; }
  return r;
}

__device__ __forceinline__ v16bf ld_b_frag(const __bf16* base, int ld, int n0, int k0) {
  const int lane = threadIdx.x & 31;
  const int n  = n0 + (lane & 15);
  const int kb = k0 + ((lane & 16) ? 8 : 0);
  const __bf16* p = base + (size_t)n * ld + kb;
  v8bf lo = *(const v8bf*)(p);
  v8bf hi = *(const v8bf*)(p + 16);
  v16bf r;
#pragma unroll
  for (int i = 0; i < 8; ++i) { r[i] = lo[i]; r[i + 8] = hi[i]; }
  return r;
}

__device__ __forceinline__ v8f wmma_bf16(v16bf a, v16bf b, v8f c) {
  return __builtin_amdgcn_wmma_f32_16x16x32_bf16(false, a, false, b, (short)0, c,
                                                 false, false);
}

// ---------------------------------------------------------------------------
// CDNA5 async global->LDS copy (ASYNCcnt path), 16B per op.
// LDS destination = low 32 bits of generic LDS pointer (aperture rule).
// ---------------------------------------------------------------------------
__device__ __forceinline__ void async_copy_b128(void* lds_ptr, const void* gptr) {
  uint32_t lds_off = (uint32_t)(uintptr_t)lds_ptr;
  uint64_t gaddr   = (uint64_t)(uintptr_t)gptr;
  asm volatile("global_load_async_to_lds_b128 %0, %1, off"
               :: "v"(lds_off), "v"(gaddr)
               : "memory");
}
__device__ __forceinline__ void wait_async0() {
  asm volatile("s_wait_asynccnt 0x0" ::: "memory");
}

// ---------------------------------------------------------------------------
// fp32 -> bf16 pack
// ---------------------------------------------------------------------------
__global__ void k_f32_to_bf16(const float* __restrict__ in, __bf16* __restrict__ out, int n) {
  int i = blockIdx.x * blockDim.x + threadIdx.x;
  if (i < n) out[i] = (__bf16)in[i];
}

// ---------------------------------------------------------------------------
// GEMM: out[M,N] = A[M,K] @ W[N,K]^T + bias
// MODE 0: f32 row-major [M,N]
// MODE 1: bf16 scattered into [B, NHEAD, S, HDIM]        (Q/K epilogue)
// MODE 2: bf16 scattered into [B, NHEAD, HDIM, S] (V^T)  (V epilogue)
// block = 256 threads = 8 waves in 4x2 grid, each wave 32x32 output tile
// ---------------------------------------------------------------------------
template <int MODE>
__global__ __launch_bounds__(256, 2)
void k_gemm_bf16_wmma(const __bf16* __restrict__ A, const __bf16* __restrict__ W,
                      const float* __restrict__ bias, float* __restrict__ outF,
                      __bf16* __restrict__ outB, int M, int N, int K) {
  const int wv   = threadIdx.x >> 5;
  const int lane = threadIdx.x & 31;
  const int m0 = blockIdx.x * 128 + (wv >> 1) * 32;
  const int n0 = blockIdx.y * 64  + (wv & 1) * 32;

  v8f acc[2][2] = {};
  for (int k0 = 0; k0 < K; k0 += 32) {
    v16bf a0 = ld_a_frag(A, K, m0,      k0);
    v16bf a1 = ld_a_frag(A, K, m0 + 16, k0);
    v16bf b0 = ld_b_frag(W, K, n0,      k0);
    v16bf b1 = ld_b_frag(W, K, n0 + 16, k0);
    acc[0][0] = wmma_bf16(a0, b0, acc[0][0]);
    acc[0][1] = wmma_bf16(a0, b1, acc[0][1]);
    acc[1][0] = wmma_bf16(a1, b0, acc[1][0]);
    acc[1][1] = wmma_bf16(a1, b1, acc[1][1]);
  }

  const int half8 = (lane & 16) ? 8 : 0;
  const int nl    = lane & 15;
#pragma unroll
  for (int i = 0; i < 2; ++i)
#pragma unroll
    for (int j = 0; j < 2; ++j)
#pragma unroll
      for (int r = 0; r < 8; ++r) {
        int m = m0 + i * 16 + r + half8;
        int n = n0 + j * 16 + nl;
        float v = acc[i][j][r] + bias[n];
        if (MODE == 0) {
          outF[(size_t)m * N + n] = v;
        } else {
          int b = m >> 11, s = m & (SS - 1);
          int h = n >> 6,  d = n & (HDIM - 1);
          if (MODE == 1)
            outB[(((size_t)b * NHEAD + h) * SS + s) * HDIM + d] = (__bf16)v;
          else // MODE == 2 : V stored pre-transposed [B,H,HDIM,S]
            outB[(((size_t)b * NHEAD + h) * HDIM + d) * SS + s] = (__bf16)v;
        }
      }
}

// ---------------------------------------------------------------------------
// RoPE on Q and K, layout [B, NHEAD, S, HDIM], bf16 storage / fp32 math.
// ---------------------------------------------------------------------------
__global__ void k_rope_bf16(__bf16* __restrict__ Q, __bf16* __restrict__ K,
                            const float* __restrict__ cosT, const float* __restrict__ sinT) {
  int t  = blockIdx.x * blockDim.x + threadIdx.x;   // B*H*S*32 threads
  int d  = t & 31;
  int s  = (t >> 5) & (SS - 1);
  int bh = t >> 16;
  size_t base = ((size_t)bh * SS + s) * HDIM;
  float c1 = cosT[s * HDIM + d],      s1 = sinT[s * HDIM + d];
  float c2 = cosT[s * HDIM + d + 32], s2 = sinT[s * HDIM + d + 32];

  float q1 = (float)Q[base + d], q2 = (float)Q[base + d + 32];
  Q[base + d]      = (__bf16)(q1 * c1 - q2 * s1);
  Q[base + d + 32] = (__bf16)(q2 * c2 + q1 * s2);

  float k1 = (float)K[base + d], k2 = (float)K[base + d + 32];
  K[base + d]      = (__bf16)(k1 * c1 - k2 * s1);
  K[base + d + 32] = (__bf16)(k2 * c2 + k1 * s2);
}

// ---------------------------------------------------------------------------
// Flash attention (non-causal). One (b,h) per blockIdx.y, 128 query rows per
// blockIdx.x, 8 waves x 16 rows. K and V^T tiles double-buffered in LDS and
// filled with global_load_async_to_lds_b128 (ASYNCcnt), overlapping the WMMA
// compute on the other buffer. Row max is reduced across lanes per tile (it
// must be lane-uniform for exp/rescale); row SUM is kept as per-lane partials
// and reduced across lanes only once, after the key loop.
// ---------------------------------------------------------------------------
#define KT   64
#define LDK  80   // 64 + 16 pad: keeps 16B alignment, breaks bank patterns
#define LDP  80
#define NKT  (SS / KT)

__global__ __launch_bounds__(256, 2)
void k_flash_attn_bf16(const __bf16* __restrict__ Q, const __bf16* __restrict__ K,
                       const __bf16* __restrict__ Vt, __bf16* __restrict__ O) {
  __shared__ __bf16 sK [2][KT][LDK];     // K tile   [key][d]
  __shared__ __bf16 sVt[2][HDIM][LDK];   // V^T tile [d][key]
  __shared__ __bf16 sP [8][16][LDP];     // per-wave P tile [row][key]

  const int wv   = threadIdx.x >> 5;
  const int lane = threadIdx.x & 31;
  const int bh   = blockIdx.y;
  const int b    = bh >> 4;
  const int h    = bh & (NHEAD - 1);
  const __bf16* Qh  = Q  + (size_t)bh * SS * HDIM;
  const __bf16* Kh  = K  + (size_t)bh * SS * HDIM;
  const __bf16* Vth = Vt + (size_t)bh * HDIM * SS;   // [d][s]

  const int q0 = blockIdx.x * 128 + wv * 16;

  // Q fragments held in registers for the whole kernel
  v16bf qf0 = ld_a_frag(Qh, HDIM, q0, 0);
  v16bf qf1 = ld_a_frag(Qh, HDIM, q0, 32);

  float rmax[8], rsum[8];   // rmax lane-uniform per 16-lane group; rsum partial
#pragma unroll
  for (int r = 0; r < 8; ++r) { rmax[r] = -1e30f; rsum[r] = 0.f; }
  v8f o[4] = {};

  const float scl = 1.4426950408889634f * 0.125f;   // log2(e)/sqrt(HDIM)

  const int t  = threadIdx.x;
  const int sr = t >> 2;          // 0..63 : tile row (key for K, d for V^T)
  const int sc = (t & 3) * 16;    // 0,16,32,48 : 16-element column chunk

  // issue async staging of tile `kt` into buffer `buf` (4 x b128 per thread)
  auto stage = [&](int buf, int kt) {
    const __bf16* kg = Kh + (size_t)(kt + sr) * HDIM + sc;
    async_copy_b128(&sK[buf][sr][sc],     kg);
    async_copy_b128(&sK[buf][sr][sc + 8], kg + 8);
    const __bf16* vg = Vth + (size_t)sr * SS + kt + sc;
    async_copy_b128(&sVt[buf][sr][sc],     vg);
    async_copy_b128(&sVt[buf][sr][sc + 8], vg + 8);
  };

  stage(0, 0);

  for (int it = 0; it < NKT; ++it) {
    const int buf = it & 1;
    wait_async0();        // this wave's async fills of `buf` have landed
    __syncthreads();      // everyone's have; also: no one still reads buf^1
    if (it + 1 < NKT) stage(buf ^ 1, (it + 1) * KT);  // overlap next copy

    // S = Q @ K^T  (16 x 64 scores per wave, 8 WMMA)
    v8f sf[4] = {};
#pragma unroll
    for (int ks = 0; ks < 2; ++ks) {
      v16bf a = ks ? qf1 : qf0;
#pragma unroll
      for (int j = 0; j < 4; ++j) {
        v16bf bfr = ld_b_frag(&sK[buf][0][0], LDK, j * 16, ks * 32);
        sf[j] = wmma_bf16(a, bfr, sf[j]);
      }
    }

    // online softmax: reduce only the row MAX across the 16-lane group
    // (offsets <=8 keep the shuffle inside the half that owns the row)
#pragma unroll
    for (int r = 0; r < 8; ++r) {
      float tmax = sf[0][r];
#pragma unroll
      for (int j = 1; j < 4; ++j) tmax = fmaxf(tmax, sf[j][r]);
#pragma unroll
      for (int off = 8; off >= 1; off >>= 1)
        tmax = fmaxf(tmax, __shfl_xor(tmax, off, 32));

      float nmax = fmaxf(rmax[r], tmax);
      float corr = exp2f((rmax[r] - nmax) * scl);   // lane-uniform
      rmax[r] = nmax;
      rsum[r] *= corr;                              // rescale per-lane partial
#pragma unroll
      for (int sub = 0; sub < 4; ++sub) o[sub][r] *= corr;

      const int mrow = r + ((lane & 16) ? 8 : 0);
#pragma unroll
      for (int j = 0; j < 4; ++j) {
        float p = exp2f((sf[j][r] - nmax) * scl);
        rsum[r] += p;                               // per-lane partial sum
        sP[wv][mrow][j * 16 + (lane & 15)] = (__bf16)p;
      }
    }

    // O += P @ V  (8 WMMA); P read back through per-wave LDS (layout convert)
#pragma unroll
    for (int ks = 0; ks < 2; ++ks) {
      v16bf pa = ld_a_frag(&sP[wv][0][0], LDP, 0, ks * 32);
#pragma unroll
      for (int sub = 0; sub < 4; ++sub) {
        v16bf vb = ld_b_frag(&sVt[buf][0][0], LDK, sub * 16, ks * 32);
        o[sub] = wmma_bf16(pa, vb, o[sub]);
      }
    }
  }

  // one-time cross-lane reduction of the row sums, then normalize + write
  const int half8 = (lane & 16) ? 8 : 0;
  const int nl    = lane & 15;
#pragma unroll
  for (int r = 0; r < 8; ++r) {
#pragma unroll
    for (int off = 8; off >= 1; off >>= 1)
      rsum[r] += __shfl_xor(rsum[r], off, 32);
    float inv = 1.0f / rsum[r];
    int s = q0 + r + half8;
#pragma unroll
    for (int sub = 0; sub < 4; ++sub) {
      int col = h * HDIM + sub * 16 + nl;
      O[((size_t)b * SS + s) * DMODEL + col] = (__bf16)(o[sub][r] * inv);
    }
  }
}

// ---------------------------------------------------------------------------
// launch
// ---------------------------------------------------------------------------
extern "C" void kernel_launch(void* const* d_in, const int* in_sizes, int n_in,
                              void* d_out, int out_size, void* d_ws, size_t ws_size,
                              hipStream_t stream) {
  (void)in_sizes; (void)n_in; (void)out_size; (void)ws_size;

  const float* x    = (const float*)d_in[0];
  const float* cosT = (const float*)d_in[1];
  const float* sinT = (const float*)d_in[2];
  const float* Wq   = (const float*)d_in[3];
  const float* bq   = (const float*)d_in[4];
  const float* Wk   = (const float*)d_in[5];
  const float* bk   = (const float*)d_in[6];
  const float* Wv   = (const float*)d_in[7];
  const float* bv   = (const float*)d_in[8];
  const float* Wo   = (const float*)d_in[9];
  const float* bo   = (const float*)d_in[10];
  float* out = (float*)d_out;

  const int NX = BB * SS * DMODEL;      // 4,194,304
  const int NW = DMODEL * DMODEL;       // 1,048,576
  const int NQ = BB * NHEAD * SS * HDIM;

  char* w = (char*)d_ws;
  size_t off = 0;
  auto alloc = [&](size_t bytes) -> __bf16* {
    __bf16* p = (__bf16*)(w + off);
    off += (bytes + 255) & ~(size_t)255;
    return p;
  };
  __bf16* xb    = alloc((size_t)NX * 2);
  __bf16* wqb   = alloc((size_t)NW * 2);
  __bf16* wkb   = alloc((size_t)NW * 2);
  __bf16* wvb   = alloc((size_t)NW * 2);
  __bf16* wob   = alloc((size_t)NW * 2);
  __bf16* Qb    = alloc((size_t)NQ * 2);
  __bf16* Kb    = alloc((size_t)NQ * 2);
  __bf16* Vtb   = alloc((size_t)NQ * 2);   // [B,H,HDIM,S]
  __bf16* attnb = alloc((size_t)NX * 2);

  // 1) pack fp32 -> bf16
  k_f32_to_bf16<<<(NX + 255) / 256, 256, 0, stream>>>(x,  xb,  NX);
  k_f32_to_bf16<<<(NW + 255) / 256, 256, 0, stream>>>(Wq, wqb, NW);
  k_f32_to_bf16<<<(NW + 255) / 256, 256, 0, stream>>>(Wk, wkb, NW);
  k_f32_to_bf16<<<(NW + 255) / 256, 256, 0, stream>>>(Wv, wvb, NW);
  k_f32_to_bf16<<<(NW + 255) / 256, 256, 0, stream>>>(Wo, wob, NW);

  // 2) QKV projections (Q/K -> [B,H,S,64]; V -> pre-transposed [B,H,64,S])
  const int M = BB * SS;  // 4096
  dim3 gg(M / 128, DMODEL / 64);
  k_gemm_bf16_wmma<1><<<gg, 256, 0, stream>>>(xb, wqb, bq, nullptr, Qb,  M, DMODEL, DMODEL);
  k_gemm_bf16_wmma<1><<<gg, 256, 0, stream>>>(xb, wkb, bk, nullptr, Kb,  M, DMODEL, DMODEL);
  k_gemm_bf16_wmma<2><<<gg, 256, 0, stream>>>(xb, wvb, bv, nullptr, Vtb, M, DMODEL, DMODEL);

  // 3) RoPE on Q, K
  k_rope_bf16<<<(BB * NHEAD * SS * 32) / 256, 256, 0, stream>>>(Qb, Kb, cosT, sinT);

  // 4) attention -> [B,S,D] bf16
  dim3 ga(SS / 128, BB * NHEAD);
  k_flash_attn_bf16<<<ga, 256, 0, stream>>>(Qb, Kb, Vtb, attnb);

  // 5) output projection -> fp32 d_out
  k_gemm_bf16_wmma<0><<<gg, 256, 0, stream>>>(attnb, wob, bo, out, nullptr, M, DMODEL, DMODEL);
}
